// LorentzLinear_75514114998724
// MI455X (gfx1250) — matrix operand order repaired and checked by reading, not compiled, not run on hardware
//
#include <hip/hip_runtime.h>
#include <cstdint>
#include <cstddef>

// ---------------------------------------------------------------------------
// CDNA5 / gfx1250 LorentzLinear:
//   out = (expmap0(x) * sign) @ expmap0(W)^T + bias
// expmap0 in fp32, downcast to bf16 (values ~sinh(45)=1.7e19 overflow fp16
// but fit bf16), GEMM via v_wmma_f32_16x16x32_bf16 with f32 accumulation.
// Global->LDS staging uses CDNA5 GLOBAL_LOAD_ASYNC_TO_LDS_B128 (ASYNCcnt),
// double-buffered so tile k+1 DMA overlaps WMMA compute on tile k.
// ---------------------------------------------------------------------------

typedef __attribute__((ext_vector_type(16))) __bf16 v16bf;
typedef __attribute__((ext_vector_type(8)))  float  v8f;

struct __align__(16) U4 { unsigned int a, b, c, d; };

union FragAB {
    v16bf v;
    U4    q[2];
};

__device__ __forceinline__ unsigned short f32_to_bf16_rne(float f) {
    unsigned int u = __float_as_uint(f);
    u += 0x7FFFu + ((u >> 16) & 1u);          // round to nearest even
    return (unsigned short)(u >> 16);
}

// ---------------------------------------------------------------------------
// Row-wise Lorentz expmap0 (k=1), one block per row, output bf16.
// ---------------------------------------------------------------------------
__global__ __launch_bounds__(256) void expmap0_bf16_kernel(
    const float* __restrict__ in, unsigned short* __restrict__ out,
    int D, int negate_first)
{
    const int row = blockIdx.x;
    const float* u = in + (size_t)row * D;
    unsigned short* o = out + (size_t)row * D;

    float part = 0.f;
    for (int i = threadIdx.x; i < D; i += 256) {
        float v = u[i];
        part = fmaf(v, (i == 0) ? -v : v, part);
    }
    #pragma unroll
    for (int off = 16; off > 0; off >>= 1)
        part += __shfl_xor(part, off, 32);

    __shared__ float red[8];
    if ((threadIdx.x & 31) == 0) red[threadIdx.x >> 5] = part;
    __syncthreads();

    const float mink = red[0] + red[1] + red[2] + red[3] +
                       red[4] + red[5] + red[6] + red[7];
    const float nomin = sqrtf(fmaxf(mink, 1e-8f));
    const float s  = sinhf(nomin) / nomin;
    const float ch = coshf(nomin);

    for (int i = threadIdx.x; i < D; i += 256) {
        float r = s * u[i];
        if (i == 0) {
            r = ch + r;
            if (negate_first) r = -r;   // fold Minkowski sign into column 0
        }
        o[i] = f32_to_bf16_rne(r);
    }
}

// ---------------------------------------------------------------------------
// bf16 WMMA GEMM:  C[m,n] = sum_k A[m,k] * B[n,k] + bias[n]
// Block: 256 threads = 8 wave32s, block tile 128x128, BK=32, double-buffered
// LDS (80B padded rows). Wave grid 2x4, each wave 64x32 = 8 accumulators.
// ---------------------------------------------------------------------------
#define BM   128
#define BN   128
#define BK   32
#define LDSS 40   // bf16 elems per LDS row: 32 data + 8 pad (80B, 16B-aligned)

// Issue one thread's 32B slice of the A and B tiles as 4 async b128 copies.
// INST_OFFSET is added to both the LDS and the global address (ISA 10.x),
// so offset:16 covers each slice's second 16B chunk with the same operands.
__device__ __forceinline__ void stage_async(
    unsigned int ldsA, unsigned int offA, unsigned long long baseA,
    unsigned int ldsB, unsigned int offB, unsigned long long baseB)
{
    asm volatile(
        "global_load_async_to_lds_b128 %0, %1, %2\n\t"
        "global_load_async_to_lds_b128 %0, %1, %2 offset:16\n\t"
        "global_load_async_to_lds_b128 %3, %4, %5\n\t"
        "global_load_async_to_lds_b128 %3, %4, %5 offset:16"
        :
        : "v"(ldsA), "v"(offA), "s"(baseA),
          "v"(ldsB), "v"(offB), "s"(baseB)
        : "memory");
}

__device__ __forceinline__ void wait_async0() {
    asm volatile("s_wait_asynccnt 0x0" ::: "memory");
}

__global__ __launch_bounds__(256) void lorentz_wmma_gemm(
    const unsigned short* __restrict__ A,
    const unsigned short* __restrict__ B,
    const float* __restrict__ bias,
    float* __restrict__ C,
    int M, int N, int K)
{
    __shared__ __align__(16) unsigned short lA[2][BM * LDSS];
    __shared__ __align__(16) unsigned short lB[2][BN * LDSS];

    const int tid = threadIdx.x;
    const int m0  = blockIdx.y * BM;
    const int n0  = blockIdx.x * BN;

    // --- async global -> LDS staging: each thread owns 32B of A and B tile ---
    const int srow = tid >> 1;
    const int scol = (tid & 1) * 16;               // bf16 elements
    const int soff = srow * LDSS + scol;

    const unsigned long long baseA = (unsigned long long)A;
    const unsigned long long baseB = (unsigned long long)B;
    unsigned int offA = (unsigned int)(((size_t)(m0 + srow) * K + scol) * 2);
    unsigned int offB = (unsigned int)(((size_t)(n0 + srow) * K + scol) * 2);

    // wave-relative LDS byte addresses (flat LDS pointer low 32 bits)
    const unsigned int ldsA0 = (unsigned int)(uintptr_t)(&lA[0][soff]);
    const unsigned int ldsB0 = (unsigned int)(uintptr_t)(&lB[0][soff]);
    const unsigned int BUFB  = BM * LDSS * 2;      // 10240B per buffer

    // prologue: DMA tile 0 into buffer 0
    stage_async(ldsA0, offA, baseA, ldsB0, offB, baseB);
    offA += BK * 2;
    offB += BK * 2;
    wait_async0();
    __syncthreads();

    const int lane = tid & 31;
    const int wave = tid >> 5;
    const int wm   = (wave >> 2) * 64;   // 0 / 64
    const int wn   = (wave & 3) * 32;    // 0 / 32 / 64 / 96
    const int l16  = lane & 15;
    const int lh   = lane >> 4;          // 0 / 1

    const v8f zacc = {0.f, 0.f, 0.f, 0.f, 0.f, 0.f, 0.f, 0.f};
    v8f acc[4][2];
    #pragma unroll
    for (int i = 0; i < 4; ++i)
        #pragma unroll
        for (int j = 0; j < 2; ++j) acc[i][j] = zacc;

    const int nk = K / BK;
    for (int kt = 0; kt < nk; ++kt) {
        const int  cur  = kt & 1;
        const bool more = (kt + 1) < nk;
        if (more) {                      // overlap DMA of tile k+1 with compute
            const unsigned int nb = (unsigned int)(cur ^ 1) * BUFB;
            stage_async(ldsA0 + nb, offA, baseA, ldsB0 + nb, offB, baseB);
            offA += BK * 2;
            offB += BK * 2;
        }

        // B fragments (contiguous k order, same as A: lanes 0-15 K=0..15, 16-31 K=16..31)
        FragAB fb0, fb1;
        {
            const unsigned short* p0 = &lB[cur][(wn +      l16) * LDSS + lh * 16];
            fb0.q[0] = *(const U4*)(p0);
            fb0.q[1] = *(const U4*)(p0 + 8);
            const unsigned short* p1 = &lB[cur][(wn + 16 + l16) * LDSS + lh * 16];
            fb1.q[0] = *(const U4*)(p1);
            fb1.q[1] = *(const U4*)(p1 + 8);
        }

        #pragma unroll
        for (int i = 0; i < 4; ++i) {
            // A fragment (lanes 0-15 K=0..15, 16-31 K=16..31, matching B)
            FragAB fa;
            const unsigned short* p = &lA[cur][(wm + i * 16 + l16) * LDSS + lh * 16];
            fa.q[0] = *(const U4*)(p);
            fa.q[1] = *(const U4*)(p + 8);
            acc[i][0] = __builtin_amdgcn_wmma_f32_16x16x32_bf16(
                false, fa.v, false, fb0.v, (short)0, acc[i][0], false, false);
            acc[i][1] = __builtin_amdgcn_wmma_f32_16x16x32_bf16(
                false, fa.v, false, fb1.v, (short)0, acc[i][1], false, false);
        }

        if (more) wait_async0();         // tile k+1 resident before barrier
        __syncthreads();
    }

    // --- epilogue: C layout (8 VGPRs: lanes0-15 M=v, lanes16-31 M=v+8) ---
    #pragma unroll
    for (int j = 0; j < 2; ++j) {
        const int col = n0 + wn + j * 16 + l16;
        const float bb = bias[col];
        #pragma unroll
        for (int i = 0; i < 4; ++i) {
            const int rbase = m0 + wm + i * 16 + lh * 8;
            #pragma unroll
            for (int v = 0; v < 8; ++v)
                C[(size_t)(rbase + v) * N + col] = acc[i][j][v] + bb;
        }
    }
}

// ---------------------------------------------------------------------------
extern "C" void kernel_launch(void* const* d_in, const int* in_sizes, int n_in,
                              void* d_out, int out_size, void* d_ws, size_t ws_size,
                              hipStream_t stream) {
    const float* x    = (const float*)d_in[0];   // (B, IN)
    const float* w    = (const float*)d_in[1];   // (OUT, IN)
    const float* bias = (const float*)d_in[2];   // (OUT,)
    float* out = (float*)d_out;

    const int N = in_sizes[2];          // OUT  = 2048
    const int D = in_sizes[1] / N;      // IN   = 2048
    const int M = in_sizes[0] / D;      // B    = 8192

    // workspace: bf16 ex (M*D) then bf16 ew (N*D) = 40 MB total
    unsigned short* exs = (unsigned short*)d_ws;
    unsigned short* ew  = exs + (size_t)M * D;

    expmap0_bf16_kernel<<<M, 256, 0, stream>>>(x, exs, D, /*negate_first=*/1);
    expmap0_bf16_kernel<<<N, 256, 0, stream>>>(w, ew,  D, /*negate_first=*/0);

    dim3 grid(N / BN, M / BM);          // (16, 64)
    lorentz_wmma_gemm<<<grid, 256, 0, stream>>>(exs, ew, bias, out, M, N, D);
}
